// BiLSTM_CRF_79972291052182
// MI455X (gfx1250) — matrix operand order repaired and checked by reading, not compiled, not run on hardware
//
#include <hip/hip_runtime.h>
#include <hip/hip_bf16.h>

// ---------------------------------------------------------------------------
// BiLSTM-CRF forward loss for gfx1250 (MI455X), wave32 + WMMA f16.
// ---------------------------------------------------------------------------
typedef __attribute__((ext_vector_type(16))) _Float16 v16h;
typedef __attribute__((ext_vector_type(8)))  float    v8f;

union V16H { v16h v; float4 f[2]; };

#define BB   64
#define TT   512
#define EE   512
#define HH   512
#define LL   17
#define BT   (BB * TT)
#define G4   (4 * HH)

// ---- WMMA helper ----------------------------------------------------------
__device__ inline v8f wmma32(v16h a, v16h b, v8f c) {
  // D = A(16x32,f16) * B(32x16,f16) + C(16x16,f32)
  return __builtin_amdgcn_wmma_f32_16x16x32_f16(
      /*neg_a=*/false, a, /*neg_b=*/false, b,
      /*c_mod=*/(short)0, c, /*reuse_a=*/false, /*reuse_b=*/false);
}

// A tile 16x32 from row-major (M,K) f16, lda % 8 == 0.
// ISA layout: lane<16 -> row=lane, K = {hi*8..+7, 16+hi*8..+7} with hi=lane>>4.
__device__ inline v16h load_a(const _Float16* __restrict__ A, int lda,
                              int m0, int k0, int lane) {
  int l = lane & 15, hi = lane >> 4;
  const _Float16* base = A + (size_t)(m0 + l) * lda + k0 + hi * 8;
  V16H u;
  u.f[0] = *reinterpret_cast<const float4*>(base);        // 8 halves
  u.f[1] = *reinterpret_cast<const float4*>(base + 16);   // 8 halves
  return u.v;
}

// B tile 32x16 where B(k,n) = Bt[n][k], Bt row-major (N,K) f16, ldb % 8 == 0.
// ISA layout: lane -> col n = lane%16, K = {hi*8..+7, 16+hi*8..+7} with hi=lane>>4.
// UNGUARDED: caller guarantees all 16 rows n0..n0+15 exist (weights are
// pre-padded on device where needed), so the K-loop stays branch-free.
__device__ inline v16h load_b(const _Float16* __restrict__ Bt, int ldb,
                              int n0, int k0, int lane) {
  int l = lane & 15, hi = lane >> 4;
  const _Float16* base = Bt + (size_t)(n0 + l) * ldb + k0 + hi * 8;
  V16H u;
  u.f[0] = *reinterpret_cast<const float4*>(base);
  u.f[1] = *reinterpret_cast<const float4*>(base + 16);
  return u.v;
}

// ---- small utility kernels ------------------------------------------------
__global__ void cvt_f32_f16(const float* __restrict__ s, _Float16* __restrict__ d, int n) {
  int i = blockIdx.x * blockDim.x + threadIdx.x;
  if (i < n) d[i] = (_Float16)s[i];
}

__global__ void cvt_pad_wout(const float* __restrict__ s, _Float16* __restrict__ d) {
  int i = blockIdx.x * blockDim.x + threadIdx.x;   // 32*1024
  int row = i >> 10, col = i & 1023;
  d[i] = (row < LL) ? (_Float16)s[row * 1024 + col] : (_Float16)0.f;
}

__global__ void embed_f16(const int* __restrict__ sent, const float* __restrict__ emb,
                          _Float16* __restrict__ x, int n) {
  int i = blockIdx.x * blockDim.x + threadIdx.x;   // n = BT*EE
  if (i >= n) return;
  int tok = i / EE, e = i - tok * EE;
  x[i] = (_Float16)emb[(size_t)sent[tok] * EE + e];
}

__global__ void zero_f16(_Float16* p, int n) {
  int i = blockIdx.x * blockDim.x + threadIdx.x;
  if (i < n) p[i] = (_Float16)0.f;
}
__global__ void zero_f32(float* p, int n) {
  int i = blockIdx.x * blockDim.x + threadIdx.x;
  if (i < n) p[i] = 0.f;
}

// ---- streaming GEMM, 16x64 per wave:  C = A(M,K)f16 * Bt(N,K)f16^T + bias -
// One wave owns 1 A tile x 4 N-tiles per K-step: 4 back-to-back WMMAs share
// the A operand (operand-reuse friendly), 1.6x less load traffic than 1-tile.
__global__ void gemm4_bias_f16f32(const _Float16* __restrict__ A,
                                  const _Float16* __restrict__ Bt,
                                  const float* __restrict__ bias,
                                  float* __restrict__ C,
                                  int M, int N, int K, int ldc) {
  int lane = threadIdx.x & 31;
  int wid  = (blockIdx.x * blockDim.x + threadIdx.x) >> 5;
  int tn = N >> 6;                        // quads of 16-wide N tiles
  if (wid >= (M >> 4) * tn) return;       // uniform per wave -> EXEC all-1s
  int m0 = (wid / tn) << 4;
  int n0 = (wid % tn) << 6;
  v8f c0 = {}, c1 = {}, c2 = {}, c3 = {};
  for (int k0 = 0; k0 < K; k0 += 32) {
    v16h a  = load_a(A, K, m0, k0, lane);
    v16h b0 = load_b(Bt, K, n0 +  0, k0, lane);
    v16h b1 = load_b(Bt, K, n0 + 16, k0, lane);
    v16h b2 = load_b(Bt, K, n0 + 32, k0, lane);
    v16h b3 = load_b(Bt, K, n0 + 48, k0, lane);
    c0 = wmma32(a, b0, c0);
    c1 = wmma32(a, b1, c1);
    c2 = wmma32(a, b2, c2);
    c3 = wmma32(a, b3, c3);
  }
  int l = lane & 15, hi = lane >> 4;
  v8f acc[4] = {c0, c1, c2, c3};
#pragma unroll
  for (int q = 0; q < 4; ++q) {
    int n = n0 + q * 16 + l;
    float bv = bias[n];
#pragma unroll
    for (int r = 0; r < 8; ++r) {
      int m = m0 + hi * 8 + r;
      C[(size_t)m * ldc + n] = acc[q][r] + bv;
    }
  }
}

// ---- small-N GEMM (emissions): 16x16 per wave, bias guarded ---------------
__global__ void gemm_bias_f16f32(const _Float16* __restrict__ A,
                                 const _Float16* __restrict__ Bt,
                                 const float* __restrict__ bias,
                                 float* __restrict__ C,
                                 int M, int N, int K, int ldc, int brows) {
  int lane = threadIdx.x & 31;
  int wid  = (blockIdx.x * blockDim.x + threadIdx.x) >> 5;
  int tn = N >> 4;
  if (wid >= (M >> 4) * tn) return;
  int m0 = (wid / tn) << 4;
  int n0 = (wid % tn) << 4;
  v8f acc = {};
  for (int k0 = 0; k0 < K; k0 += 32) {
    v16h a = load_a(A, K, m0, k0, lane);
    v16h b = load_b(Bt, K, n0, k0, lane);   // Bt pre-padded with zero rows
    acc = wmma32(a, b, acc);
  }
  int l = lane & 15, hi = lane >> 4;
  int n = n0 + l;
  float bv = (n < brows) ? bias[n] : 0.f;
#pragma unroll
  for (int r = 0; r < 8; ++r) {
    int m = m0 + hi * 8 + r;
    C[(size_t)m * ldc + n] = acc[r] + bv;
  }
}

// ---- fused recurrent step: g = Gx[t] + h*Whh^T ; gates ; c,h update -------
// One wave = 16 batch rows x 16 hidden units, all 4 gates (A reused 4x).
__global__ void lstm_step_kernel(const float* __restrict__ Gx,      // (BT, 4H) f32, bias folded in
                                 const _Float16* __restrict__ Whh,  // (4H, H) f16 row-major
                                 const _Float16* __restrict__ hin,  // (B, H) f16
                                 _Float16* __restrict__ hout,       // (B, H) f16
                                 float* __restrict__ c,             // (B, H) f32
                                 _Float16* __restrict__ xcat,       // (BT, 2H) f16 layer output
                                 int t, int dcat) {
  int lane = threadIdx.x & 31;
  int wid  = (blockIdx.x * blockDim.x + threadIdx.x) >> 5;  // 0..127
  int m0 = (wid >> 5) << 4;   // batch tile
  int u0 = (wid & 31) << 4;   // unit tile
  v8f a0 = {}, a1 = {}, a2 = {}, a3 = {};
  for (int k0 = 0; k0 < HH; k0 += 32) {
    v16h a  = load_a(hin, HH, m0, k0, lane);
    v16h b0 = load_b(Whh, HH, 0 * HH + u0, k0, lane);
    v16h b1 = load_b(Whh, HH, 1 * HH + u0, k0, lane);
    v16h b2 = load_b(Whh, HH, 2 * HH + u0, k0, lane);
    v16h b3 = load_b(Whh, HH, 3 * HH + u0, k0, lane);
    a0 = wmma32(a, b0, a0);
    a1 = wmma32(a, b1, a1);
    a2 = wmma32(a, b2, a2);
    a3 = wmma32(a, b3, a3);
  }
  int l = lane & 15, hi = lane >> 4;
  int u = u0 + l;
#pragma unroll
  for (int r = 0; r < 8; ++r) {
    int m = m0 + hi * 8 + r;                    // batch index
    size_t xrow = (size_t)m * TT + t;
    const float* g = Gx + xrow * G4 + u;
    float gi = a0[r] + g[0 * HH];
    float gf = a1[r] + g[1 * HH];
    float gg = a2[r] + g[2 * HH];
    float go = a3[r] + g[3 * HH];
    float si = 1.f / (1.f + __expf(-gi));
    float sf = 1.f / (1.f + __expf(-gf));
    float so = 1.f / (1.f + __expf(-go));
    float tg = tanhf(gg);
    size_t ci = (size_t)m * HH + u;
    float cn = sf * c[ci] + si * tg;
    c[ci] = cn;
    float hv = so * tanhf(cn);
    hout[ci] = (_Float16)hv;
    xcat[xrow * (2 * HH) + dcat + u] = (_Float16)hv;
  }
}

// ---- CRF: one wave per batch; lane j holds score[j] -----------------------
__global__ void crf_batch_kernel(const float* __restrict__ emis,   // (BT, 32) f32
                                 const unsigned char* __restrict__ mask,
                                 const int* __restrict__ labels,
                                 const float* __restrict__ st,
                                 const float* __restrict__ en,
                                 const float* __restrict__ trans,
                                 float* __restrict__ res) {
  int b = blockIdx.x;
  int j = threadIdx.x;                   // 0..31, tags 0..16 live
  const float NEG = -1e30f;
  size_t base = (size_t)b * TT;
  float score = (j < LL) ? st[j] + emis[base * 32 + j] : NEG;
  float trow[LL];
  if (j < LL)
    for (int i = 0; i < LL; ++i) trow[i] = trans[i * LL + j];
  for (int t = 1; t < TT; ++t) {
    int mt = mask[base + t];
    float vmax = NEG;
    float vs[LL];
    for (int i = 0; i < LL; ++i) {
      float si = __shfl(score, i, 32);
      float v = (j < LL) ? si + trow[i] : NEG;
      vs[i] = v;
      vmax = fmaxf(vmax, v);
    }
    float s = 0.f;
    for (int i = 0; i < LL; ++i) s += __expf(vs[i] - vmax);
    float nxt = vmax + __logf(s) + emis[(base + t) * 32 + j];
    if (mt && j < LL) score = nxt;
  }
  // logZ = logsumexp_j(score + end)
  float v = (j < LL) ? score + en[j] : NEG;
  float m = v;
  for (int off = 16; off; off >>= 1) m = fmaxf(m, __shfl_xor(m, off, 32));
  float e = (j < LL) ? __expf(v - m) : 0.f;
  for (int off = 16; off; off >>= 1) e += __shfl_xor(e, off, 32);
  float logZ = m + __logf(e);
  // numerator (gold path score), lanes stride over time
  float local = 0.f;
  int cnt = 0;
  for (int t = j; t < TT; t += 32) {
    int mt = mask[base + t];
    cnt += mt;
    if (t >= 1 && mt) {
      int lp = labels[base + t - 1], lc = labels[base + t];
      local += trans[lp * LL + lc] + emis[(base + t) * 32 + lc];
    }
  }
  for (int off = 16; off; off >>= 1) {
    local += __shfl_xor(local, off, 32);
    cnt += __shfl_xor(cnt, off, 32);
  }
  if (j == 0) {
    int l0 = labels[base];
    float num = local + st[l0] + emis[base * 32 + l0];
    num += en[labels[base + cnt - 1]];
    res[b] = num - logZ;
  }
}

__global__ void crf_final_kernel(const float* __restrict__ res, float* __restrict__ out) {
  int j = threadIdx.x;
  float v = res[j] + res[j + 32];
  for (int off = 16; off; off >>= 1) v += __shfl_xor(v, off, 32);
  if (j == 0) out[0] = -v / (float)BB;
}

// ---------------------------------------------------------------------------
extern "C" void kernel_launch(void* const* d_in, const int* in_sizes, int n_in,
                              void* d_out, int out_size, void* d_ws, size_t ws_size,
                              hipStream_t stream) {
  (void)in_sizes; (void)n_in; (void)out_size; (void)ws_size;
  const int*           d_sent  = (const int*)d_in[0];
  const unsigned char* d_mask  = (const unsigned char*)d_in[1];  // numpy bool = 1 byte
  const int*           d_lab   = (const int*)d_in[2];
  const float*         d_emb   = (const float*)d_in[3];
  // d_in[4..15] = 2 layers x (Wif, Whf, bf, Wib, Whb, bb)
  const float* W_out   = (const float*)d_in[16];
  const float* b_out   = (const float*)d_in[17];
  const float* start_t = (const float*)d_in[18];
  const float* end_t   = (const float*)d_in[19];
  const float* trans   = (const float*)d_in[20];

  // workspace carve-up (256B aligned)
  char* ws = (char*)d_ws;
  size_t off = 0;
  auto take = [&](size_t bytes) { size_t r = off; off = (off + bytes + 255) & ~(size_t)255; return r; };
  _Float16* x1   = (_Float16*)(ws + take((size_t)BT * EE * 2));          // embedded f16
  _Float16* x2   = (_Float16*)(ws + take((size_t)BT * 1024 * 2));        // layer0 concat
  _Float16* x3   = (_Float16*)(ws + take((size_t)BT * 1024 * 2));        // layer1 concat
  float*    Gx   = (float*)   (ws + take((size_t)BT * G4 * 4));          // input projection
  _Float16* wA16 = (_Float16*)(ws + take((size_t)G4 * 1024 * 2));        // Wih f16 (max K=1024)
  _Float16* wB16 = (_Float16*)(ws + take((size_t)G4 * HH * 2));          // Whh f16
  _Float16* wO16 = (_Float16*)(ws + take((size_t)32 * 1024 * 2));        // W_out padded f16
  _Float16* h16  = (_Float16*)(ws + take((size_t)2 * BB * HH * 2));      // h ping-pong
  float*    cbuf = (float*)   (ws + take((size_t)BB * HH * 4));          // cell state
  float*    emis = (float*)   (ws + take((size_t)BT * 32 * 4));          // emissions (padded 32)
  float*    res  = (float*)   (ws + take((size_t)BB * 4));               // per-batch num-logZ

  // 1) embedding lookup -> f16
  {
    int n = BT * EE;
    embed_f16<<<n / 256, 256, 0, stream>>>(d_sent, d_emb, x1, n);
  }

  // 2) two bidirectional LSTM layers
  for (int layer = 0; layer < 2; ++layer) {
    const _Float16* xin = (layer == 0) ? x1 : x2;
    _Float16* xout = (layer == 0) ? x2 : x3;
    int Kin = (layer == 0) ? EE : 1024;
    for (int dir = 0; dir < 2; ++dir) {
      const float* Wih = (const float*)d_in[4 + layer * 6 + dir * 3 + 0];
      const float* Whh = (const float*)d_in[4 + layer * 6 + dir * 3 + 1];
      const float* bb  = (const float*)d_in[4 + layer * 6 + dir * 3 + 2];
      cvt_f32_f16<<<(G4 * Kin) / 256, 256, 0, stream>>>(Wih, wA16, G4 * Kin);
      cvt_f32_f16<<<(G4 * HH) / 256, 256, 0, stream>>>(Whh, wB16, G4 * HH);
      // Gx = X * Wih^T + b   (32768 x 2048, K = Kin), 16x64 per wave
      {
        int waves = (BT / 16) * (G4 / 64);
        gemm4_bias_f16f32<<<waves * 32 / 256, 256, 0, stream>>>(
            xin, wA16, bb, Gx, BT, G4, Kin, G4);
      }
      zero_f16<<<(2 * BB * HH) / 256, 256, 0, stream>>>(h16, 2 * BB * HH);
      zero_f32<<<(BB * HH) / 256, 256, 0, stream>>>(cbuf, BB * HH);
      for (int s = 0; s < TT; ++s) {
        int t = dir ? (TT - 1 - s) : s;
        const _Float16* hin = h16 + (size_t)(s & 1) * (BB * HH);
        _Float16* hout = h16 + (size_t)((s + 1) & 1) * (BB * HH);
        lstm_step_kernel<<<16, 256, 0, stream>>>(Gx, wB16, hin, hout, cbuf,
                                                 xout, t, dir * HH);
      }
    }
  }

  // 3) emissions = x3 * W_out^T + b_out  (N padded 17 -> 32, zero rows)
  cvt_pad_wout<<<(32 * 1024) / 256, 256, 0, stream>>>(W_out, wO16);
  {
    int waves = (BT / 16) * (32 / 16);
    gemm_bias_f16f32<<<waves * 32 / 256, 256, 0, stream>>>(
        x3, wO16, b_out, emis, BT, 32, 1024, 32, LL);
  }

  // 4) CRF loss
  crf_batch_kernel<<<BB, 32, 0, stream>>>(emis, d_mask, d_lab, start_t, end_t, trans, res);
  crf_final_kernel<<<1, 32, 0, stream>>>(res, (float*)d_out);
}
